// NodeEdgeBlock_8203387536041
// MI455X (gfx1250) — matrix operand order repaired and checked by reading, not compiled, not run on hardware
//
#include <hip/hip_runtime.h>
#include <hip/hip_bf16.h>
#include <math.h>

// ---------------------------------------------------------------------------
// NodeEdgeBlock forward for MI455X (gfx1250, wave32, WMMA 16x16x32 f16).
// bs=4, n=512, dx=256, de=64, dy=64, nh=8, df=32
// ---------------------------------------------------------------------------

typedef __attribute__((ext_vector_type(16))) _Float16 v16h;
typedef __attribute__((ext_vector_type(8)))  float    v8f;

#define DEV __device__ __forceinline__

DEV v8f wmma16(v16h a, v16h b, v8f c) {
  // D = A(16x32 f16) x B(32x16 f16) + C(16x16 f32)
  return __builtin_amdgcn_wmma_f32_16x16x32_f16(
      /*neg_a=*/false, a, /*neg_b=*/false, b,
      /*c_mod=*/(short)0, c, /*reuse_a=*/false, /*reuse_b=*/false);
}

DEV int lane_id() { return (int)(threadIdx.x & 31); }

// A tile 16x32 for f16 WMMA from row-major f32, leading dim ld (floats).
// Layout used here (contiguous k, same order as B): lanes 0-15 row M=lane hold K=0..15; lanes 16-31 hold
// K=16..31 for the same rows.
DEV v16h load_A_f32(const float* base, int ld) {
  int l = lane_id();
  const float* p = base + (size_t)(l & 15) * ld + ((l >> 4) << 4);
  v16h a;
#pragma unroll
  for (int i = 0; i < 8; ++i) { a[i] = (_Float16)p[i]; a[i + 8] = (_Float16)p[i + 8]; }
  return a;
}

DEV v16h load_A_f16(const _Float16* base, int ld) {
  int l = lane_id();
  const _Float16* p = base + (size_t)(l & 15) * ld + ((l >> 4) << 4);
  v16h a;
#pragma unroll
  for (int i = 0; i < 8; ++i) { a[i] = p[i]; a[i + 8] = p[i + 8]; }
  return a;
}

// B tile 32x16 where B[k][n] = src[n*ld + k] (src row-major N x K).
// Layout used here (contiguous k, same order as A): lane column n = lane&15; lanes 0-15 K=0..15, lanes 16-31 K=16..31.
DEV v16h load_B_f32(const float* base, int ld) {
  int l = lane_id();
  const float* p = base + (size_t)(l & 15) * ld + ((l >> 4) << 4);
  v16h b;
#pragma unroll
  for (int i = 0; i < 16; ++i) b[i] = (_Float16)p[i];
  return b;
}

DEV v16h load_B_f16(const _Float16* base, int ld) {
  int l = lane_id();
  const _Float16* p = base + (size_t)(l & 15) * ld + ((l >> 4) << 4);
  v16h b;
#pragma unroll
  for (int i = 0; i < 16; ++i) b[i] = p[i];
  return b;
}

// ---------------------------------------------------------------------------
// K1: Q/K/V GEMMs (2048x256)@(256x256)^T with WMMA. z selects {Q,K,V}.
// Q,K stored f16 row-major; V stored transposed Vt[b][h][d][n] (f16).
// ---------------------------------------------------------------------------
__global__ void k_qkv(const float* __restrict__ X, const float* __restrict__ X1,
                      const float* __restrict__ qW, const float* __restrict__ qb,
                      const float* __restrict__ kW, const float* __restrict__ kb,
                      const float* __restrict__ vW, const float* __restrict__ vb,
                      const unsigned char* __restrict__ mask,
                      _Float16* __restrict__ Qh, _Float16* __restrict__ Kh,
                      _Float16* __restrict__ Vt) {
  int mt = blockIdx.x, nt = blockIdx.y, mat = blockIdx.z;
  const float* src  = (mat == 0) ? X1 : X;
  const float* W    = (mat == 0) ? qW : (mat == 1) ? kW : vW;
  const float* bias = (mat == 0) ? qb : (mat == 1) ? kb : vb;
  int row0 = mt * 16;
  v8f acc = {};
#pragma unroll
  for (int kc = 0; kc < 8; ++kc) {
    v16h a = load_A_f32(src + (size_t)row0 * 256 + kc * 32, 256);
    v16h b = load_B_f32(W + (size_t)(nt * 16) * 256 + kc * 32, 256);
    acc = wmma16(a, b, acc);
  }
  int l = lane_id(), col = l & 15, hi = l >> 4;
  int f = nt * 16 + col;
  float bv = bias[f];
#pragma unroll
  for (int j = 0; j < 8; ++j) {
    int row = row0 + j + 8 * hi;       // 0..2047 = b*512+nn
    int bidx = row >> 9, nn = row & 511;
    float v = (acc[j] + bv) * (mask[bidx * 512 + nn] ? 1.0f : 0.0f);
    if (mat == 0)      Qh[(size_t)row * 256 + f] = (_Float16)v;
    else if (mat == 1) Kh[(size_t)row * 256 + f] = (_Float16)v;
    else {
      int h = f >> 5, d = f & 31;
      Vt[(((size_t)(bidx * 8 + h) * 32 + d) << 9) + nn] = (_Float16)v;
    }
  }
}

// ---------------------------------------------------------------------------
// K2 (fused edge kernel), one wave per (b, q-tile16, k-tile16):
//   scores (8 WMMA) -> E/GRN projections (stacked-weight WMMAs) -> FiLM(y) ->
//   alpha gating -> eout 8->64 expansion (4 WMMA/row) -> d_out newE,
//   plus modulated scores Yw[b][h][q][k] for the softmax pass.
// ---------------------------------------------------------------------------
__global__ void k_edge(const _Float16* __restrict__ Qh, const _Float16* __restrict__ Kh,
                       const float* __restrict__ E, const float* __restrict__ GRN,
                       const float* __restrict__ emulW, const float* __restrict__ emulb,
                       const float* __restrict__ eaddW, const float* __restrict__ eaddb,
                       const float* __restrict__ gmulW, const float* __restrict__ gmulb,
                       const float* __restrict__ gaddW, const float* __restrict__ gaddb,
                       const float* __restrict__ eoutW, const float* __restrict__ eoutb,
                       const float* __restrict__ ye12, const float* __restrict__ t_init,
                       float* __restrict__ Yw, float* __restrict__ newE_out) {
  __shared__ float smemY[8][16][16];   // [h][q_local][k_local]
  __shared__ float smemN[16][8];       // [k_local][h]
  int kt = blockIdx.x, qt = blockIdx.y, b = blockIdx.z;
  int l = lane_id(), col = l & 15, hi = l >> 4, h8 = l & 7;

  // alpha from cosine schedule (ratio form; ac[0] normalization cancels)
  int idx = (int)(t_init[b] * 1000.0f);
  const float s = 0.008f, PI = 3.14159265358979323846f;
  float x0 = 1002.0f * (float)idx / 1001.0f;
  float x1 = 1002.0f * (float)(idx + 1) / 1001.0f;
  float c0 = cosf(0.5f * PI * (x0 / 1002.0f + s) / (1.0f + s));
  float c1 = cosf(0.5f * PI * (x1 / 1002.0f + s) / (1.0f + s));
  float beta = 1.0f - (c1 * c1) / (c0 * c0);
  beta = fminf(fmaxf(beta, 0.0f), 0.9999f);
  float alpha = 1.0f - beta;

  float ye1 = ye12[b * 16 + h8];
  float ye2 = ye12[b * 16 + 8 + h8];

  // ---- attention scores: one WMMA per head (K = df = 32 exactly) ----
  const float scale = 0.17677669529663687f; // 1/sqrt(32)
#pragma unroll
  for (int h = 0; h < 8; ++h) {
    v16h a  = load_A_f16(Qh + ((size_t)(b * 512 + qt * 16)) * 256 + h * 32, 256);
    v16h bm = load_B_f16(Kh + ((size_t)(b * 512 + kt * 16)) * 256 + h * 32, 256);
    v8f acc = {};
    acc = wmma16(a, bm, acc);
#pragma unroll
    for (int j = 0; j < 8; ++j) smemY[h][j + 8 * hi][col] = acc[j] * scale;
  }
  __syncthreads();

  // ---- stacked B operands: cols 0-7 = mul weights, 8-15 = add weights ----
  v16h BE[2], BG[2];
  {
    const float* rm = ((col < 8) ? emulW : eaddW) + h8 * 64;
    const float* rg = ((col < 8) ? gmulW : gaddW) + h8 * 64;
    int kb = hi * 16;
#pragma unroll
    for (int c = 0; c < 2; ++c)
#pragma unroll
      for (int i = 0; i < 16; ++i) {
        BE[c][i] = (_Float16)rm[c * 32 + kb + i];
        BG[c][i] = (_Float16)rg[c * 32 + kb + i];
      }
  }
  float ebias = (col < 8) ? emulb[h8] : eaddb[h8];
  float gbias = (col < 8) ? gmulb[h8] : gaddb[h8];

  // ---- eout B operands (4 N-chunks of 16; K=8 real + zero pad to 32) ----
  v16h Bout[4];
#pragma unroll
  for (int c = 0; c < 4; ++c) {
#pragma unroll
    for (int i = 0; i < 16; ++i) Bout[c][i] = (_Float16)0.0f;
    if (hi == 0) {
      const float* r = eoutW + (size_t)(c * 16 + col) * 8;
#pragma unroll
      for (int i = 0; i < 8; ++i) Bout[c][i] = (_Float16)r[i];
    }
  }

  size_t ebase = (((size_t)(b * 512 + qt * 16)) * 512 + kt * 16) * 64;
#pragma unroll 1
  for (int g = 0; g < 16; ++g) {  // g = q row within tile
    int q = qt * 16 + g;
    const float* Eg = E + ebase + (size_t)g * (512 * 64);
    const float* Gg = GRN + ebase + (size_t)g * (512 * 64);
    if (g + 1 < 16) {  // prefetch next group's edge rows (global_prefetch_b8)
      __builtin_prefetch(Eg + 512 * 64, 0, 1);
      __builtin_prefetch(Gg + 512 * 64, 0, 1);
    }
    // projections: rows = 16 k-neighbors, K = 64 edge features (2 chunks)
    v8f accE = {}, accG = {};
    accE = wmma16(load_A_f32(Eg, 64),      BE[0], accE);
    accE = wmma16(load_A_f32(Eg + 32, 64), BE[1], accE);
    accG = wmma16(load_A_f32(Gg, 64),      BG[0], accG);
    accG = wmma16(load_A_f32(Gg + 32, 64), BG[1], accG);
#pragma unroll
    for (int j = 0; j < 8; ++j) {
      int kl = j + 8 * hi;
      float e  = accE[j] + ebias;
      float gg = accG[j] + gbias;
      float eo = __shfl_xor(e, 8, 32);
      float go = __shfl_xor(gg, 8, 32);
      float e1 = (l & 8) ? eo : e;
      float e2 = (l & 8) ? e : eo;
      float g1 = (l & 8) ? go : gg;
      float g2 = (l & 8) ? gg : go;
      float ym = smemY[h8][g][kl] * (e1 + 1.0f) + e2;  // E-modulated score
      float ne = ye1 + (ye2 + 1.0f) * ym;              // FiLM(y)
      if (Gg[(size_t)kl * 64] != 0.0f)                 // GRN gate (zm==false)
        ne = alpha * ne * g1 + (1.0f - alpha) * ne + alpha * g2;
      if ((l & 8) == 0) {
        Yw[(((size_t)(b * 8 + h8) * 512 + q) << 9) + kt * 16 + kl] = ym;
        smemN[kl][h8] = ne;
      }
    }
    __syncthreads();
    // eout: newE8(16x8) @ eoutW^T(8x64) as 4 WMMAs (zero-padded K)
    v16h Aout;
#pragma unroll
    for (int i = 0; i < 16; ++i) Aout[i] = (_Float16)0.0f;
    if (hi == 0) {
#pragma unroll
      for (int i = 0; i < 8; ++i) Aout[i] = (_Float16)smemN[col][i];
    }
#pragma unroll
    for (int c = 0; c < 4; ++c) {
      v8f accO = {};
      accO = wmma16(Aout, Bout[c], accO);
      int f = c * 16 + col;
      float ob = eoutb[f];
#pragma unroll
      for (int j = 0; j < 8; ++j) {
        int kl = j + 8 * hi;
        newE_out[(((size_t)(b * 512 + q) << 9) + kt * 16 + kl) * 64 + f] = accO[j] + ob;
      }
    }
    __syncthreads();
  }
}

// ---------------------------------------------------------------------------
// K3: softmax over k (512 contiguous f32) per (b,h,q); writes f16 attn.
// ---------------------------------------------------------------------------
__global__ void k_softmax(const float* __restrict__ Yw, _Float16* __restrict__ At) {
  int q = blockIdx.x, h = blockIdx.y, b = blockIdx.z;
  int l = (int)threadIdx.x;
  const float* row = Yw + (((size_t)(b * 8 + h) * 512 + q) << 9);
  float vals[16];
  float mx = -INFINITY;
#pragma unroll
  for (int i = 0; i < 16; ++i) { vals[i] = row[l + 32 * i]; mx = fmaxf(mx, vals[i]); }
#pragma unroll
  for (int o = 16; o > 0; o >>= 1) mx = fmaxf(mx, __shfl_xor(mx, o, 32));
  float sum = 0.0f;
#pragma unroll
  for (int i = 0; i < 16; ++i) { vals[i] = expf(vals[i] - mx); sum += vals[i]; }
#pragma unroll
  for (int o = 16; o > 0; o >>= 1) sum += __shfl_xor(sum, o, 32);
  float inv = 1.0f / sum;
  _Float16* orow = At + (((size_t)(b * 8 + h) * 512 + q) << 9);
#pragma unroll
  for (int i = 0; i < 16; ++i) orow[l + 32 * i] = (_Float16)(vals[i] * inv);
}

// ---------------------------------------------------------------------------
// K4: weighted_V = attn @ V per (b,h), fused FiLM (yx1,yx2) -> f16 Xin.
// ---------------------------------------------------------------------------
__global__ void k_av(const _Float16* __restrict__ At, const _Float16* __restrict__ Vt,
                     const float* __restrict__ yx12, _Float16* __restrict__ Xin) {
  int qt = blockIdx.x, dt = blockIdx.y, bh = blockIdx.z;
  int b = bh >> 3, h = bh & 7;
  const _Float16* abase = At + (((size_t)bh << 9) + qt * 16) * 512;
  const _Float16* vbase = Vt + (((size_t)bh * 32 + dt * 16) << 9);
  v8f acc = {};
#pragma unroll 1
  for (int kc = 0; kc < 16; ++kc) {
    v16h a  = load_A_f16(abase + kc * 32, 512);
    v16h bm = load_B_f16(vbase + kc * 32, 512);
    acc = wmma16(a, bm, acc);
  }
  int l = lane_id(), col = l & 15, hi = l >> 4;
  int f = h * 32 + dt * 16 + col;
  float x1 = yx12[b * 512 + f];
  float x2 = yx12[b * 512 + 256 + f];
#pragma unroll
  for (int j = 0; j < 8; ++j) {
    int q = qt * 16 + j + 8 * hi;
    Xin[((size_t)(b * 512 + q)) * 256 + f] = (_Float16)(x1 + (x2 + 1.0f) * acc[j]);
  }
}

// ---------------------------------------------------------------------------
// K5: newX = (Xin @ xoutW^T + b) * mask  (WMMA GEMM) -> d_out
// ---------------------------------------------------------------------------
__global__ void k_xout(const _Float16* __restrict__ Xin, const float* __restrict__ xoutW,
                       const float* __restrict__ xoutb, const unsigned char* __restrict__ mask,
                       float* __restrict__ outX) {
  int mt = blockIdx.x, nt = blockIdx.y;
  v8f acc = {};
#pragma unroll
  for (int kc = 0; kc < 8; ++kc) {
    v16h a = load_A_f16(Xin + (size_t)mt * 16 * 256 + kc * 32, 256);
    v16h b = load_B_f32(xoutW + (size_t)(nt * 16) * 256 + kc * 32, 256);
    acc = wmma16(a, b, acc);
  }
  int l = lane_id(), col = l & 15, hi = l >> 4;
  int f = nt * 16 + col;
  float bv = xoutb[f];
#pragma unroll
  for (int j = 0; j < 8; ++j) {
    int row = mt * 16 + j + 8 * hi;
    int bidx = row >> 9, nn = row & 511;
    outX[(size_t)row * 256 + f] = (acc[j] + bv) * (mask[bidx * 512 + nn] ? 1.0f : 0.0f);
  }
}

// ---------------------------------------------------------------------------
// Small per-batch y projections: ye1,ye2 (8 each), yx1,yx2 (256 each).
// ---------------------------------------------------------------------------
__global__ void k_yproj(const float* __restrict__ y,
                        const float* __restrict__ yeaddW, const float* __restrict__ yeaddb,
                        const float* __restrict__ yemulW, const float* __restrict__ yemulb,
                        const float* __restrict__ yxaddW, const float* __restrict__ yxaddb,
                        const float* __restrict__ yxmulW, const float* __restrict__ yxmulb,
                        float* __restrict__ ye12, float* __restrict__ yx12) {
  int b = blockIdx.x, t = (int)threadIdx.x;  // 256 threads
  __shared__ float ysh[64];
  if (t < 64) ysh[t] = y[b * 64 + t];
  __syncthreads();
  float s1 = yxaddb[t], s2 = yxmulb[t];
  for (int i = 0; i < 64; ++i) {
    s1 += ysh[i] * yxaddW[t * 64 + i];
    s2 += ysh[i] * yxmulW[t * 64 + i];
  }
  yx12[b * 512 + t] = s1;
  yx12[b * 512 + 256 + t] = s2;
  if (t < 8) {
    float a1 = yeaddb[t], a2 = yemulb[t];
    for (int i = 0; i < 64; ++i) {
      a1 += ysh[i] * yeaddW[t * 64 + i];
      a2 += ysh[i] * yemulW[t * 64 + i];
    }
    ye12[b * 16 + t] = a1;
    ye12[b * 16 + 8 + t] = a2;
  }
}

// ---------------------------------------------------------------------------
// E pool-stats stage 1: per (b,q) partial sum/sumsq/min/max over k, per channel.
// ---------------------------------------------------------------------------
__global__ void k_estats1(const float* __restrict__ E, float* __restrict__ part) {
  int q = blockIdx.x, b = blockIdx.y, c = (int)threadIdx.x;  // 64 threads
  const float* base = E + (((size_t)(b * 512 + q)) << 9) * 64 + c;
  float sum = 0.0f, sq = 0.0f, mn = INFINITY, mx = -INFINITY;
  for (int k = 0; k < 512; ++k) {
    float v = base[(size_t)k * 64];
    sum += v; sq += v * v; mn = fminf(mn, v); mx = fmaxf(mx, v);
  }
  float* p = part + (((size_t)(b * 512 + q)) * 64 + c) * 4;
  p[0] = sum; p[1] = sq; p[2] = mn; p[3] = mx;
}

__global__ void k_estats2(const float* __restrict__ part, float* __restrict__ estats) {
  int b = blockIdx.x, c = (int)threadIdx.x;  // 64 threads
  float sum = 0.0f, sq = 0.0f, mn = INFINITY, mx = -INFINITY;
  for (int q = 0; q < 512; ++q) {
    const float* p = part + (((size_t)(b * 512 + q)) * 64 + c) * 4;
    sum += p[0]; sq += p[1]; mn = fminf(mn, p[2]); mx = fmaxf(mx, p[3]);
  }
  const float Nn = 512.0f * 512.0f;
  float mean = sum / Nn;
  float var = (sq - Nn * mean * mean) / (Nn - 1.0f);
  estats[b * 256 + c]       = mean;
  estats[b * 256 + 64 + c]  = mn;
  estats[b * 256 + 128 + c] = mx;
  estats[b * 256 + 192 + c] = sqrtf(fmaxf(var, 0.0f));
}

__global__ void k_xstats(const float* __restrict__ X, float* __restrict__ xstats) {
  int b = blockIdx.x, c = (int)threadIdx.x;  // 256 threads
  float sum = 0.0f, sq = 0.0f, mn = INFINITY, mx = -INFINITY;
  for (int nn = 0; nn < 512; ++nn) {
    float v = X[((size_t)(b * 512 + nn)) * 256 + c];
    sum += v; sq += v * v; mn = fminf(mn, v); mx = fmaxf(mx, v);
  }
  float mean = sum / 512.0f;
  float var = (sq - 512.0f * mean * mean) / 511.0f;
  xstats[b * 1024 + c]       = mean;
  xstats[b * 1024 + 256 + c] = mn;
  xstats[b * 1024 + 512 + c] = mx;
  xstats[b * 1024 + 768 + c] = sqrtf(fmaxf(var, 0.0f));
}

// ---------------------------------------------------------------------------
// new_y MLP (tiny): yy + xy(stats(X)) + ey(stats(E)) -> gelu -> linear.
// ---------------------------------------------------------------------------
__global__ void k_newy(const float* __restrict__ y,
                       const float* __restrict__ yyW, const float* __restrict__ yyb,
                       const float* __restrict__ xyW, const float* __restrict__ xyb,
                       const float* __restrict__ eyW, const float* __restrict__ eyb,
                       const float* __restrict__ yout1W, const float* __restrict__ yout1b,
                       const float* __restrict__ yout2W, const float* __restrict__ yout2b,
                       const float* __restrict__ xstats, const float* __restrict__ estats,
                       float* __restrict__ outy) {
  int b = blockIdx.x, o = (int)threadIdx.x;  // 64 threads
  __shared__ float ny[64], gg[64];
  float s = yyb[o] + xyb[o] + eyb[o];
  for (int i = 0; i < 64; ++i)   s += y[b * 64 + i] * yyW[o * 64 + i];
  for (int i = 0; i < 1024; ++i) s += xstats[b * 1024 + i] * xyW[o * 1024 + i];
  for (int i = 0; i < 256; ++i)  s += estats[b * 256 + i] * eyW[o * 256 + i];
  ny[o] = s;
  __syncthreads();
  float t = yout1b[o];
  for (int i = 0; i < 64; ++i) t += ny[i] * yout1W[o * 64 + i];
  t = 0.5f * t * (1.0f + erff(t * 0.70710678118654752f));  // exact gelu
  gg[o] = t;
  __syncthreads();
  float u = yout2b[o];
  for (int i = 0; i < 64; ++i) u += gg[i] * yout2W[o * 64 + i];
  outy[b * 64 + o] = u;
}

// ---------------------------------------------------------------------------
extern "C" void kernel_launch(void* const* d_in, const int* in_sizes, int n_in,
                              void* d_out, int out_size, void* d_ws, size_t ws_size,
                              hipStream_t stream) {
  (void)in_sizes; (void)n_in; (void)out_size; (void)ws_size;
  const float* X      = (const float*)d_in[0];
  const float* X1     = (const float*)d_in[1];
  const float* E      = (const float*)d_in[2];
  const float* y      = (const float*)d_in[3];
  const float* t_init = (const float*)d_in[4];
  const float* GRN    = (const float*)d_in[5];
  const unsigned char* mask = (const unsigned char*)d_in[6];
  const float* qW = (const float*)d_in[7];   const float* qb = (const float*)d_in[8];
  const float* kW = (const float*)d_in[9];   const float* kb = (const float*)d_in[10];
  const float* vW = (const float*)d_in[11];  const float* vb = (const float*)d_in[12];
  const float* emulW = (const float*)d_in[13]; const float* emulb = (const float*)d_in[14];
  const float* eaddW = (const float*)d_in[15]; const float* eaddb = (const float*)d_in[16];
  const float* gmulW = (const float*)d_in[17]; const float* gmulb = (const float*)d_in[18];
  const float* gaddW = (const float*)d_in[19]; const float* gaddb = (const float*)d_in[20];
  const float* yeaddW = (const float*)d_in[21]; const float* yeaddb = (const float*)d_in[22];
  const float* yemulW = (const float*)d_in[23]; const float* yemulb = (const float*)d_in[24];
  const float* yxaddW = (const float*)d_in[25]; const float* yxaddb = (const float*)d_in[26];
  const float* yxmulW = (const float*)d_in[27]; const float* yxmulb = (const float*)d_in[28];
  const float* yyW = (const float*)d_in[29];  const float* yyb = (const float*)d_in[30];
  const float* xyW = (const float*)d_in[31];  const float* xyb = (const float*)d_in[32];
  const float* eyW = (const float*)d_in[33];  const float* eyb = (const float*)d_in[34];
  const float* xoutW = (const float*)d_in[35]; const float* xoutb = (const float*)d_in[36];
  const float* eoutW = (const float*)d_in[37]; const float* eoutb = (const float*)d_in[38];
  const float* yout1W = (const float*)d_in[39]; const float* yout1b = (const float*)d_in[40];
  const float* yout2W = (const float*)d_in[41]; const float* yout2b = (const float*)d_in[42];

  // workspace layout (bytes)
  char* ws = (char*)d_ws;
  _Float16* Qh  = (_Float16*)(ws + 0);                       // 1 MB
  _Float16* Kh  = (_Float16*)(ws + 1048576u);                // 1 MB
  _Float16* Vt  = (_Float16*)(ws + 2097152u);                // 1 MB
  float*    Yw  = (float*)   (ws + 3145728u);                // 33.5 MB
  _Float16* At  = (_Float16*)(ws + 36700160u);               // 16.8 MB
  _Float16* Xin = (_Float16*)(ws + 53477376u);               // 1 MB
  float*    ye12   = (float*)(ws + 54525952u);               // 256 B
  float*    yx12   = (float*)(ws + 54526208u);               // 8 KB
  float*    xstats = (float*)(ws + 54534400u);               // 16 KB
  float*    estats = (float*)(ws + 54550784u);               // 4 KB
  float*    epart  = (float*)(ws + 54554880u);               // 2 MB

  float* outX = (float*)d_out;                       // (4,512,256)
  float* outE = outX + (size_t)4 * 512 * 256;        // (4,512,512,64)
  float* outy = outE + (size_t)4 * 512 * 512 * 64;   // (4,64)

  k_yproj<<<4, 256, 0, stream>>>(y, yeaddW, yeaddb, yemulW, yemulb,
                                 yxaddW, yxaddb, yxmulW, yxmulb, ye12, yx12);
  k_xstats<<<4, 256, 0, stream>>>(X, xstats);
  k_estats1<<<dim3(512, 4), 64, 0, stream>>>(E, epart);
  k_estats2<<<4, 64, 0, stream>>>(epart, estats);

  k_qkv<<<dim3(128, 16, 3), 32, 0, stream>>>(X, X1, qW, qb, kW, kb, vW, vb,
                                             mask, Qh, Kh, Vt);
  k_edge<<<dim3(32, 32, 4), 32, 0, stream>>>(Qh, Kh, E, GRN,
                                             emulW, emulb, eaddW, eaddb,
                                             gmulW, gmulb, gaddW, gaddb,
                                             eoutW, eoutb, ye12, t_init,
                                             Yw, outE);
  k_softmax<<<dim3(512, 8, 4), 32, 0, stream>>>(Yw, At);
  k_av<<<dim3(32, 2, 32), 32, 0, stream>>>(At, Vt, yx12, Xin);
  k_xout<<<dim3(128, 16), 32, 0, stream>>>(Xin, xoutW, xoutb, mask, outX);
  k_newy<<<4, 64, 0, stream>>>(y, yyW, yyb, xyW, xyb, eyW, eyb,
                               yout1W, yout1b, yout2W, yout2b,
                               xstats, estats, outy);
}
